// PointerNetwork_64304250356209
// MI455X (gfx1250) — matrix operand order repaired and checked by reading, not compiled, not run on hardware
//
#include <hip/hip_runtime.h>
#include <hip/hip_bf16.h>

typedef _Float16 f16;
typedef __attribute__((ext_vector_type(16))) _Float16 v16h;
typedef __attribute__((ext_vector_type(8)))  _Float16 v8h;
typedef __attribute__((ext_vector_type(8)))  float    v8f;

#define BATCH 4096
#define SEQL  10
#define HID   512
#define H3    1536

// ---------------------------------------------------------------- activations
__device__ __forceinline__ float sigmoidf_(float x) {
    x = fminf(fmaxf(x, -30.f), 30.f);
    return 1.0f / (1.0f + __expf(-x));
}
__device__ __forceinline__ float tanhf_(float x) {
    x = fminf(fmaxf(x, -15.f), 15.f);
    float e = __expf(-2.0f * x);
    return (1.0f - e) / (1.0f + e);
}

// ---------------------------------------------------------------- WMMA frags
// A fragment: 16x32 (MxK) f16 tile at `A` (tile origin), row stride lda halves.
// lane<16: M=lane, K in {0..15}; lane>=16: M=lane-16, K in {16..31}
__device__ __forceinline__ v16h load_a_frag(const f16* __restrict__ A, int lda, int lane) {
    int m    = lane & 15;
    int koff = (lane >> 4) * 16;
    const f16* p = A + (size_t)m * lda + koff;
    return *(const v16h*)(p);
}
// B fragment: 32x16 (KxN). Weight stored row-major (N,K): lane holds column
// N=lane%16, K contiguous range [ (lane/16)*16, +16 ) -> one 32B load.
__device__ __forceinline__ v16h load_b_frag(const f16* __restrict__ Wt, int ldb, int lane) {
    int n    = lane & 15;
    int koff = (lane >> 4) * 16;
    return *(const v16h*)(Wt + (size_t)n * ldb + koff);
}

// ---------------------------------------------------------------- WMMA GEMM
// C[M,N] = A[M,K] (f16, row-major) * Wt[N,K]^T (f16, row-major) [+ bias[N]]
// Block: 256 thr = 8 waves arranged 2(M) x 4(N). Block tile 128x256.
// Wave tile 64x64 = 4x4 fragments (8 frag loads per 16 WMMAs). K step 32.
template<bool HAS_BIAS>
__global__ __launch_bounds__(256) void gemm_f16_wmma(
    const f16* __restrict__ A, const f16* __restrict__ Wt,
    const float* __restrict__ bias, float* __restrict__ C,
    int M, int N, int K)
{
    const int lane = threadIdx.x & 31;
    const int wid  = threadIdx.x >> 5;
    const int wm   = wid >> 2;          // 0..1
    const int wn   = wid & 3;           // 0..3
    const int row0 = blockIdx.y * 128 + wm * 64;
    const int col0 = blockIdx.x * 256 + wn * 64;

    v8f zero = {0.f,0.f,0.f,0.f,0.f,0.f,0.f,0.f};
    v8f acc[4][4];
#pragma unroll
    for (int i = 0; i < 4; ++i)
#pragma unroll
        for (int j = 0; j < 4; ++j) acc[i][j] = zero;

    for (int k0 = 0; k0 < K; k0 += 32) {
        v16h a[4], b[4];
#pragma unroll
        for (int i = 0; i < 4; ++i)
            a[i] = load_a_frag(A + (size_t)(row0 + 16 * i) * K + k0, K, lane);
#pragma unroll
        for (int j = 0; j < 4; ++j)
            b[j] = load_b_frag(Wt + (size_t)(col0 + 16 * j) * K + k0, K, lane);
#pragma unroll
        for (int i = 0; i < 4; ++i)
#pragma unroll
            for (int j = 0; j < 4; ++j)
                acc[i][j] = __builtin_amdgcn_wmma_f32_16x16x32_f16(
                    false, a[i], false, b[j], (short)0, acc[i][j], false, false);
    }

    // Epilogue: C/D layout -> VGPR v holds M = v + 8*(lane/16), N = lane%16
    const int n  = lane & 15;
    const int mo = (lane >> 4) * 8;
#pragma unroll
    for (int i = 0; i < 4; ++i) {
#pragma unroll
        for (int j = 0; j < 4; ++j) {
            int cb = col0 + 16 * j + n;
            float bv = HAS_BIAS ? bias[cb] : 0.0f;
#pragma unroll
            for (int v = 0; v < 8; ++v) {
                int r = row0 + 16 * i + mo + v;
                C[(size_t)r * N + cb] = acc[i][j][v] + bv;
            }
        }
    }
}

// ---------------------------------------------------------------- small kernels
__global__ void f32_to_f16_kernel(const float* __restrict__ src, f16* __restrict__ dst, int n) {
    int i = blockIdx.x * blockDim.x + threadIdx.x;
    if (i < n) dst[i] = (f16)src[i];
}

__global__ void zero_h_kernel(float* __restrict__ h, f16* __restrict__ h16, int n) {
    int i = blockIdx.x * blockDim.x + threadIdx.x;
    if (i < n) { h[i] = 0.0f; h16[i] = (f16)0.0f; }
}

// Wf[j][i] = sum_h enc_wih[j,h]*emb_w[h,i];  bf[j] = enc_bih[j] + sum_h enc_wih[j,h]*emb_b[h]
__global__ void fuse_embed_kernel(
    const float* __restrict__ enc_wih, const float* __restrict__ emb_w,
    const float* __restrict__ emb_b,   const float* __restrict__ enc_bih,
    float* __restrict__ Wf, float* __restrict__ bf)
{
    int j = blockIdx.x * blockDim.x + threadIdx.x;
    if (j >= H3) return;
    float acc[8] = {0.f,0.f,0.f,0.f,0.f,0.f,0.f,0.f};
    float accb = 0.f;
    for (int h = 0; h < HID; ++h) {
        float w = enc_wih[(size_t)j * HID + h];
#pragma unroll
        for (int i = 0; i < 8; ++i) acc[i] += w * emb_w[h * 8 + i];
        accb += w * emb_b[h];
    }
#pragma unroll
    for (int i = 0; i < 8; ++i) Wf[j * 8 + i] = acc[i];
    bf[j] = accb + enc_bih[j];
}

// Encoder gate: xg computed on the fly from fused (IN=8 -> 3H) weight.
// hg already contains h@Whh^T + bhh. Updates h (in place) and writes f16 copy.
__global__ void enc_gate_kernel(
    const float* __restrict__ items, int t,
    const float* __restrict__ Wf, const float* __restrict__ bf,
    const float* __restrict__ hg,
    float* __restrict__ h, f16* __restrict__ hout16)
{
    int idx = blockIdx.x * blockDim.x + threadIdx.x;
    if (idx >= BATCH * HID) return;
    int b = idx >> 9, j = idx & 511;
    const float* it = items + ((size_t)b * SEQL + t) * 8;
    float xg[3];
#pragma unroll
    for (int g = 0; g < 3; ++g) {
        const float* row = Wf + (size_t)(g * HID + j) * 8;
        float s = bf[g * HID + j];
#pragma unroll
        for (int i = 0; i < 8; ++i) s += it[i] * row[i];
        xg[g] = s;
    }
    const float* hgb = hg + (size_t)b * H3;
    float r = sigmoidf_(xg[0] + hgb[j]);
    float z = sigmoidf_(xg[1] + hgb[j + HID]);
    float nn = tanhf_(xg[2] + r * hgb[j + 2 * HID]);
    float hn = (1.0f - z) * nn + z * h[idx];
    h[idx] = hn;
    hout16[idx] = (f16)hn;
}

// Decoder gate: xg precomputed once (xg_dec = dec_in@dec_wih^T + dec_bih).
__global__ void dec_gate_kernel(
    const float* __restrict__ xg, const float* __restrict__ hg,
    float* __restrict__ h, f16* __restrict__ hout16)
{
    int idx = blockIdx.x * blockDim.x + threadIdx.x;
    if (idx >= BATCH * HID) return;
    int b = idx >> 9, j = idx & 511;
    const float* xgb = xg + (size_t)b * H3;
    const float* hgb = hg + (size_t)b * H3;
    float r = sigmoidf_(xgb[j]           + hgb[j]);
    float z = sigmoidf_(xgb[j + HID]     + hgb[j + HID]);
    float nn = tanhf_(xgb[j + 2 * HID] + r * hgb[j + 2 * HID]);
    float hn = (1.0f - z) * nn + z * h[idx];
    h[idx] = hn;
    hout16[idx] = (f16)hn;
}

// out_seq[b][t][l] = sum_w tanh(enc_w3[l,b,w] + a4[b,w]) * v2[w]
// One wave per (b,l) pair; 8 waves per block.
__global__ __launch_bounds__(256) void out_seq_kernel(
    const float* __restrict__ encw3, const float* __restrict__ a4,
    const float* __restrict__ v2, float* __restrict__ out, int t)
{
    int wg   = blockIdx.x * (blockDim.x >> 5) + (threadIdx.x >> 5);
    int lane = threadIdx.x & 31;
    if (wg >= BATCH * SEQL) return;
    int b = wg / SEQL;
    int l = wg - b * SEQL;
    const float* e = encw3 + ((size_t)l * BATCH + b) * HID;
    const float* a = a4 + (size_t)b * HID;
    float s = 0.f;
    for (int w = lane; w < HID; w += 32) s += tanhf_(e[w] + a[w]) * v2[w];
#pragma unroll
    for (int off = 16; off; off >>= 1) s += __shfl_xor(s, off, 32);
    if (lane == 0) out[(size_t)b * (SEQL * SEQL) + t * SEQL + l] = s;
}

// out_ori[b][t][j] = bori[j] + sum_k h[b,k] * wori[j, 512 + k]   (he = hi = 0)
__global__ __launch_bounds__(256) void out_ori_kernel(
    const float* __restrict__ h, const float* __restrict__ wori,
    const float* __restrict__ bori, float* __restrict__ out, int t)
{
    int wg   = blockIdx.x * (blockDim.x >> 5) + (threadIdx.x >> 5);
    int lane = threadIdx.x & 31;
    if (wg >= BATCH * 6) return;
    int b = wg / 6;
    int j = wg - b * 6;
    const float* hb = h + (size_t)b * HID;
    const float* w  = wori + (size_t)j * H3 + HID;
    float s = 0.f;
    for (int k = lane; k < HID; k += 32) s += hb[k] * w[k];
#pragma unroll
    for (int off = 16; off; off >>= 1) s += __shfl_xor(s, off, 32);
    if (lane == 0)
        out[(size_t)BATCH * SEQL * SEQL + (size_t)b * (SEQL * 6) + t * 6 + j] = s + bori[j];
}

// ---------------------------------------------------------------- launcher
extern "C" void kernel_launch(void* const* d_in, const int* in_sizes, int n_in,
                              void* d_out, int out_size, void* d_ws, size_t ws_size,
                              hipStream_t stream) {
    const float* items   = (const float*)d_in[0];
    const float* dec_in  = (const float*)d_in[1];
    const float* emb_w   = (const float*)d_in[2];
    const float* emb_b   = (const float*)d_in[3];
    const float* enc_wih = (const float*)d_in[4];
    const float* enc_whh = (const float*)d_in[5];
    const float* enc_bih = (const float*)d_in[6];
    const float* enc_bhh = (const float*)d_in[7];
    const float* dec_wih = (const float*)d_in[8];
    const float* dec_whh = (const float*)d_in[9];
    const float* dec_bih = (const float*)d_in[10];
    const float* dec_bhh = (const float*)d_in[11];
    const float* w3      = (const float*)d_in[14];
    const float* w4      = (const float*)d_in[15];
    const float* v2      = (const float*)d_in[20];
    const float* wori    = (const float*)d_in[23];
    const float* bori    = (const float*)d_in[24];
    float* out = (float*)d_out;

    // ---- workspace carve-out (256B aligned)
    char* ws = (char*)d_ws;
    size_t off = 0;
    auto take = [&](size_t bytes) -> char* {
        char* p = ws + off;
        off = (off + bytes + 255) & ~(size_t)255;
        return p;
    };
    float* Wf        = (float*)take((size_t)H3 * 8 * 4);
    float* bf        = (float*)take((size_t)H3 * 4);
    f16* enc_whh16   = (f16*)take((size_t)H3 * HID * 2);
    f16* dec_whh16   = (f16*)take((size_t)H3 * HID * 2);
    f16* dec_wih16   = (f16*)take((size_t)H3 * HID * 2);
    f16* w3_16       = (f16*)take((size_t)HID * HID * 2);
    f16* w4_16       = (f16*)take((size_t)HID * HID * 2);
    f16* dec_in16    = (f16*)take((size_t)BATCH * HID * 2);
    f16* h16_zero    = (f16*)take((size_t)BATCH * HID * 2);
    f16* h_dec16     = (f16*)take((size_t)BATCH * HID * 2);
    f16* enc_out16   = (f16*)take((size_t)SEQL * BATCH * HID * 2);
    float* h_f32     = (float*)take((size_t)BATCH * HID * 4);
    float* hg        = (float*)take((size_t)BATCH * H3 * 4);
    float* xg_dec    = (float*)take((size_t)BATCH * H3 * 4);
    float* enc_w3    = (float*)take((size_t)SEQL * BATCH * HID * 4);
    float* a4        = (float*)take((size_t)BATCH * HID * 4);
    (void)ws_size; (void)in_sizes; (void)n_in; (void)out_size;

    const int TPB = 256;
    auto blocks = [](size_t n, int t) { return (unsigned)((n + t - 1) / t); };

    // ---- precompute: fused embed weight + f16 weight copies
    fuse_embed_kernel<<<blocks(H3, TPB), TPB, 0, stream>>>(enc_wih, emb_w, emb_b, enc_bih, Wf, bf);
    f32_to_f16_kernel<<<blocks((size_t)H3 * HID, TPB), TPB, 0, stream>>>(enc_whh, enc_whh16, H3 * HID);
    f32_to_f16_kernel<<<blocks((size_t)H3 * HID, TPB), TPB, 0, stream>>>(dec_whh, dec_whh16, H3 * HID);
    f32_to_f16_kernel<<<blocks((size_t)H3 * HID, TPB), TPB, 0, stream>>>(dec_wih, dec_wih16, H3 * HID);
    f32_to_f16_kernel<<<blocks((size_t)HID * HID, TPB), TPB, 0, stream>>>(w3, w3_16, HID * HID);
    f32_to_f16_kernel<<<blocks((size_t)HID * HID, TPB), TPB, 0, stream>>>(w4, w4_16, HID * HID);
    f32_to_f16_kernel<<<blocks((size_t)BATCH * HID, TPB), TPB, 0, stream>>>(dec_in, dec_in16, BATCH * HID);
    zero_h_kernel<<<blocks((size_t)BATCH * HID, TPB), TPB, 0, stream>>>(h_f32, h16_zero, BATCH * HID);

    // ---- encoder GRU: 10 dependent steps
    for (int t = 0; t < SEQL; ++t) {
        const f16* hin = (t == 0) ? h16_zero : enc_out16 + (size_t)(t - 1) * BATCH * HID;
        gemm_f16_wmma<true><<<dim3(H3 / 256, BATCH / 128), TPB, 0, stream>>>(
            hin, enc_whh16, enc_bhh, hg, BATCH, H3, HID);
        enc_gate_kernel<<<blocks((size_t)BATCH * HID, TPB), TPB, 0, stream>>>(
            items, t, Wf, bf, hg, h_f32, enc_out16 + (size_t)t * BATCH * HID);
    }

    // ---- enc_w3 = enc_out @ w3^T  (M = L*B = 40960)
    gemm_f16_wmma<false><<<dim3(HID / 256, (SEQL * BATCH) / 128), TPB, 0, stream>>>(
        enc_out16, w3_16, nullptr, enc_w3, SEQL * BATCH, HID, HID);

    // ---- xg_dec = dec_in @ dec_wih^T + dec_bih  (fixed across decoder steps)
    gemm_f16_wmma<true><<<dim3(H3 / 256, BATCH / 128), TPB, 0, stream>>>(
        dec_in16, dec_wih16, dec_bih, xg_dec, BATCH, H3, HID);

    // ---- decoder: 10 dependent steps (h0 = encoder final state, already in h_f32)
    for (int t = 0; t < SEQL; ++t) {
        const f16* hin = (t == 0) ? enc_out16 + (size_t)(SEQL - 1) * BATCH * HID : h_dec16;
        gemm_f16_wmma<true><<<dim3(H3 / 256, BATCH / 128), TPB, 0, stream>>>(
            hin, dec_whh16, dec_bhh, hg, BATCH, H3, HID);
        dec_gate_kernel<<<blocks((size_t)BATCH * HID, TPB), TPB, 0, stream>>>(
            xg_dec, hg, h_f32, h_dec16);
        gemm_f16_wmma<false><<<dim3(HID / 256, BATCH / 128), TPB, 0, stream>>>(
            h_dec16, w4_16, nullptr, a4, BATCH, HID, HID);
        out_seq_kernel<<<(BATCH * SEQL) / 8, TPB, 0, stream>>>(enc_w3, a4, v2, out, t);
        out_ori_kernel<<<(BATCH * 6) / 8, TPB, 0, stream>>>(h_f32, wori, bori, out, t);
    }
}
